// PointNet2Encoder_55637006353053
// MI455X (gfx1250) — compile-verified, matrix operand order checked by reading, not run on hardware
//
#include <hip/hip_runtime.h>
#include <hip/hip_bf16.h>

// Problem constants
#define B_ 8
#define N_ 16384
#define M_ 512     // SAMPLES
#define H_ 128     // HIDDEN
#define C_ 256     // CGEO
#define PTILE 128  // points per block in fused kernel
#define NWAVE 8    // waves per block (wave32)
#define NCHUNK 32  // g-reduction chunks per batch

typedef __attribute__((ext_vector_type(16))) __bf16        v16bf;
typedef __attribute__((ext_vector_type(8)))  float         v8f;
typedef __attribute__((ext_vector_type(4)))  unsigned int  u32x4;

union BF16x16 {
  v16bf v;
  __bf16 e[16];
  u32x4 q[2];
};

// ---------------------------------------------------------------------------
// Kernel 1: farthest point sampling. 1 block per batch, 1024 threads,
// 16 points per thread held in registers. Sequential 512-step argmax.
// ---------------------------------------------------------------------------
__global__ __launch_bounds__(1024)
void fps_kernel(const float* __restrict__ pts, int* __restrict__ idx) {
  const int b = blockIdx.x;
  const int t = threadIdx.x;
  const float* p = pts + (size_t)b * N_ * 3;
  float px[16], py[16], pz[16], dist[16];
#pragma unroll
  for (int i = 0; i < 16; ++i) {
    int n = t + i * 1024;
    px[i] = p[n * 3 + 0];
    py[i] = p[n * 3 + 1];
    pz[i] = p[n * 3 + 2];
    dist[i] = 1e10f;
  }
  __shared__ float sc[3];
  __shared__ float wv[32];
  __shared__ int   wi[32];
  __shared__ int   sfar;
  if (t == 0) sfar = 0;
  __syncthreads();
  const int lane = t & 31, wid = t >> 5;
  for (int it = 0; it < M_; ++it) {
    if (t == 0) {
      int f = sfar;
      idx[b * M_ + it] = f;            // reference emits far BEFORE update
      sc[0] = p[f * 3 + 0];
      sc[1] = p[f * 3 + 1];
      sc[2] = p[f * 3 + 2];
    }
    __syncthreads();
    float cx = sc[0], cy = sc[1], cz = sc[2];
    float best = -1.0f; int bi = 0;
#pragma unroll
    for (int i = 0; i < 16; ++i) {
      float dx = px[i] - cx, dy = py[i] - cy, dz = pz[i] - cz;
      float d = dx * dx + dy * dy + dz * dz;
      float nd = fminf(dist[i], d);
      dist[i] = nd;
      if (nd > best) { best = nd; bi = t + i * 1024; } // strict > keeps lowest idx
    }
#pragma unroll
    for (int off = 16; off >= 1; off >>= 1) {
      float ov = __shfl_xor(best, off, 32);
      int   oi = __shfl_xor(bi, off, 32);
      if (ov > best || (ov == best && oi < bi)) { best = ov; bi = oi; }
    }
    if (lane == 0) { wv[wid] = best; wi[wid] = bi; }
    __syncthreads();
    if (t == 0) {
      float bb = wv[0]; int bj = wi[0];
      for (int k = 1; k < 32; ++k)
        if (wv[k] > bb || (wv[k] == bb && wi[k] < bj)) { bb = wv[k]; bj = wi[k]; }
      sfar = bj;
    }
  }
}

// ---------------------------------------------------------------------------
// Kernel 2: transpose W2a/W2b to column-major bf16 (K contiguous per column)
// so WMMA B-fragments are contiguous 16B chunks per lane.
// ---------------------------------------------------------------------------
__global__ __launch_bounds__(256)
void prep_weights(const float* __restrict__ W2a, const float* __restrict__ W2b,
                  __bf16* __restrict__ w2a_cm, __bf16* __restrict__ w2b_cm) {
  int i = blockIdx.x * blockDim.x + threadIdx.x;
  if (i < H_ * C_) {                 // W2a: [128 k][256 o] -> [256 o][128 k]
    int k = i / C_, o = i % C_;
    w2a_cm[o * H_ + k] = (__bf16)W2a[i];
  }
  if (i < C_ * C_) {                 // W2b: [256 k][256 o] -> [256 o][256 k]
    int k = i / C_, o = i % C_;
    w2b_cm[o * C_ + k] = (__bf16)W2b[i];
  }
}

// ---------------------------------------------------------------------------
// Kernel 3: anchors = MLP1(pts[idx]) for the 4096 sampled points only
// (x = MLP1(pts) is only consumed through the gather). One block per sample.
// Stores anchors column-major bf16 [b][h][m] and sampled xyz [b][dim][m].
// ---------------------------------------------------------------------------
__global__ __launch_bounds__(128)
void anchors_kernel(const float* __restrict__ pts, const int* __restrict__ idx,
                    const float* __restrict__ W1a, const float* __restrict__ b1a,
                    const float* __restrict__ W1b, const float* __restrict__ b1b,
                    __bf16* __restrict__ anch_cm, float* __restrict__ sxyz) {
  const int blk = blockIdx.x;
  const int b = blk >> 9, m = blk & 511;
  const int t = threadIdx.x;
  const int pi = idx[b * M_ + m];
  const float* p = pts + ((size_t)b * N_ + pi) * 3;
  const float x = p[0], y = p[1], z = p[2];
  if (t < 3) sxyz[(b * 3 + t) * M_ + m] = p[t];
  __shared__ float h1[H_];
  float h = x * W1a[0 * H_ + t] + y * W1a[1 * H_ + t] + z * W1a[2 * H_ + t] + b1a[t];
  h1[t] = fmaxf(h, 0.0f);
  __syncthreads();
  float acc = b1b[t];
  for (int k = 0; k < H_; ++k) acc += h1[k] * W1b[k * H_ + t];
  acc = fmaxf(acc, 0.0f);
  anch_cm[((size_t)b * H_ + t) * M_ + m] = (__bf16)acc;
}

// ---------------------------------------------------------------------------
// Kernel 4: fused  dists -> softmax -> w@anchors -> MLP2 -> F_geo
// Block = 128 points (8 waves, each wave owns a 16-row M strip).
// All WMMA B operands are staged through a 64KB shared LDS region (s_B),
// loaded cooperatively once per block and consumed by all 8 waves:
//   GEMM1: anchors[512x128]  in two 64KB K-halves
//   GEMM2: W2a[128x256]      whole (64KB)
//   GEMM3: W2b[256x256]      in two 64KB K-halves
// This cuts per-block L2 reads from ~2.5MB (per-wave refetch) to 320KB.
// Softmax normalization is deferred to a row scale of the GEMM1 accumulator.
// ---------------------------------------------------------------------------
__global__ __launch_bounds__(256)
void fused_kernel(const float* __restrict__ pts,
                  const __bf16* __restrict__ anch_cm,
                  const float* __restrict__ sxyz,
                  const __bf16* __restrict__ w2a_cm, const float* __restrict__ b2a,
                  const __bf16* __restrict__ w2b_cm, const float* __restrict__ b2b,
                  float* __restrict__ out) {
  const int blk = blockIdx.x;
  const int b = blk >> 7;              // 128 tiles per batch
  const int tile = blk & 127;
  const int p0 = tile * PTILE;
  const int t = threadIdx.x;
  const int w = t >> 5, lane = t & 31;
  const int half = lane >> 4, l16 = lane & 15;

  __shared__ float s_sx[M_], s_sy[M_], s_sz[M_];
  __shared__ __align__(16) __bf16 s_B[32768];             // 64KB shared B region
  __shared__ __align__(16) __bf16 s_stage[NWAVE][16 * C_]; // 8KB per wave
  __shared__ float s_inv[NWAVE][16];

  const __bf16* anchB = anch_cm + (size_t)b * H_ * M_;
  __builtin_prefetch(anchB + (size_t)t * M_, 0, 0);   // warm L2

  for (int i = t; i < M_; i += 256) {
    s_sx[i] = sxyz[(b * 3 + 0) * M_ + i];
    s_sy[i] = sxyz[(b * 3 + 1) * M_ + i];
    s_sz[i] = sxyz[(b * 3 + 2) * M_ + i];
  }

  // this lane's point == A-matrix row l16 of this wave's strip
  const int prow = p0 + w * 16 + l16;
  const float px = pts[((size_t)b * N_ + prow) * 3 + 0];
  const float py = pts[((size_t)b * N_ + prow) * 3 + 1];
  const float pz = pts[((size_t)b * N_ + prow) * 3 + 2];

  // ---- GEMM1: X1[16x128] = softmax_w[16x512] @ anchors[512x128] ----
  v8f acc1[8];
#pragma unroll
  for (int n = 0; n < 8; ++n)
#pragma unroll
    for (int r = 0; r < 8; ++r) acc1[n][r] = 0.0f;

  float esum = 0.0f;
  for (int c = 0; c < 2; ++c) {            // K halves: m in [c*256, c*256+256)
    __syncthreads();                       // prior users of s_B done (also covers s_sx)
    // cooperative load: anchors half -> s_B[h][256], 16B per copy
#pragma unroll
    for (int j = t; j < 4096; j += 256) {
      int h = j >> 5, pos = (j & 31) * 8;
      *(u32x4*)(s_B + h * 256 + pos) = *(const u32x4*)(anchB + h * M_ + c * 256 + pos);
    }
    __syncthreads();
    for (int k2 = 0; k2 < 8; ++k2) {
      const int k = c * 8 + k2;
      // Build A fragment in ISA layout: lane half 0 -> K {0..7,16..23}+32k,
      // half 1 -> K {8..15,24..31}+32k. exp(-d) <= 1 so no max-subtraction;
      // normalization applied post-GEMM via 1/sum.
      BF16x16 a;
#pragma unroll
      for (int i = 0; i < 8; ++i) {
        int s = 32 * k + 8 * half + i;
        float dx = px - s_sx[s], dy = py - s_sy[s], dz = pz - s_sz[s];
        float d = sqrtf(dx * dx + dy * dy + dz * dz);
        float e = __expf(-d);
        esum += e;
        a.e[i] = (__bf16)e;
        int s2 = s + 16;
        dx = px - s_sx[s2]; dy = py - s_sy[s2]; dz = pz - s_sz[s2];
        d = sqrtf(dx * dx + dy * dy + dz * dz);
        e = __expf(-d);
        esum += e;
        a.e[8 + i] = (__bf16)e;
      }
#pragma unroll
      for (int n = 0; n < 8; ++n) {
        BF16x16 bb;
        const __bf16* colp = s_B + (n * 16 + l16) * 256 + 32 * k2 + 8 * half;
        bb.q[0] = *(const u32x4*)(colp);
        bb.q[1] = *(const u32x4*)(colp + 16);
        acc1[n] = __builtin_amdgcn_wmma_f32_16x16x32_bf16(
            false, a.v, false, bb.v, (short)0, acc1[n], false, false);
      }
    }
  }
  esum += __shfl_xor(esum, 16, 32);          // lane pair covers all 512 samples
  if (half == 0) s_inv[w][l16] = 1.0f / esum;  // wave-internal; DS in-order
  float rinv[8];
#pragma unroll
  for (int r = 0; r < 8; ++r) rinv[r] = s_inv[w][r + 8 * half];

  // stage X1 row-major bf16 [16][128] (per-wave region)
  __bf16* stg = &s_stage[w][0];
#pragma unroll
  for (int n = 0; n < 8; ++n)
#pragma unroll
    for (int r = 0; r < 8; ++r) {
      float v = acc1[n][r] * rinv[r];
      stg[(r + 8 * half) * H_ + n * 16 + l16] = (__bf16)v;
    }

  // ---- GEMM2: H[16x256] = relu(X1[16x128] @ W2a[128x256] + b2a) ----
  __syncthreads();                       // all waves done with anchors in s_B
#pragma unroll
  for (int j = t; j < 4096; j += 256) {  // load W2a whole: s_B[o][128]
    int o = j >> 4, pos = (j & 15) * 8;
    *(u32x4*)(s_B + o * H_ + pos) = *(const u32x4*)(w2a_cm + o * H_ + pos);
  }
  __syncthreads();

  v8f acc2[16];
#pragma unroll
  for (int n = 0; n < 16; ++n)
#pragma unroll
    for (int r = 0; r < 8; ++r) acc2[n][r] = 0.0f;

  for (int k = 0; k < 4; ++k) {
    BF16x16 a;
    const __bf16* ap = stg + l16 * H_ + 32 * k + 8 * half;
    a.q[0] = *(const u32x4*)(ap);
    a.q[1] = *(const u32x4*)(ap + 16);
#pragma unroll
    for (int n = 0; n < 16; ++n) {
      BF16x16 bb;
      const __bf16* colp = s_B + (n * 16 + l16) * H_ + 32 * k + 8 * half;
      bb.q[0] = *(const u32x4*)(colp);
      bb.q[1] = *(const u32x4*)(colp + 16);
      acc2[n] = __builtin_amdgcn_wmma_f32_16x16x32_bf16(
          false, a.v, false, bb.v, (short)0, acc2[n], false, false);
    }
  }
#pragma unroll
  for (int n = 0; n < 16; ++n) {
    float bias = b2a[n * 16 + l16];
#pragma unroll
    for (int r = 0; r < 8; ++r) {
      float v = fmaxf(acc2[n][r] + bias, 0.0f);
      stg[(r + 8 * half) * C_ + n * 16 + l16] = (__bf16)v;   // H [16][256]
    }
  }

  // ---- GEMM3: F[16x256] = relu(H[16x256] @ W2b[256x256] + b2b) -> out ----
  v8f acc3[16];
#pragma unroll
  for (int n = 0; n < 16; ++n)
#pragma unroll
    for (int r = 0; r < 8; ++r) acc3[n][r] = 0.0f;

  for (int c = 0; c < 2; ++c) {            // K halves: k in [c*128, c*128+128)
    __syncthreads();                       // previous s_B users done
#pragma unroll
    for (int j = t; j < 4096; j += 256) {  // load W2b half: s_B[o][128]
      int o = j >> 4, pos = (j & 15) * 8;
      *(u32x4*)(s_B + o * H_ + pos) = *(const u32x4*)(w2b_cm + o * C_ + c * H_ + pos);
    }
    __syncthreads();
    for (int k2 = 0; k2 < 4; ++k2) {
      const int k = c * 4 + k2;
      BF16x16 a;
      const __bf16* ap = stg + l16 * C_ + 32 * k + 8 * half;
      a.q[0] = *(const u32x4*)(ap);
      a.q[1] = *(const u32x4*)(ap + 16);
#pragma unroll
      for (int n = 0; n < 16; ++n) {
        BF16x16 bb;
        const __bf16* colp = s_B + (n * 16 + l16) * H_ + 32 * k2 + 8 * half;
        bb.q[0] = *(const u32x4*)(colp);
        bb.q[1] = *(const u32x4*)(colp + 16);
        acc3[n] = __builtin_amdgcn_wmma_f32_16x16x32_bf16(
            false, a.v, false, bb.v, (short)0, acc3[n], false, false);
      }
    }
  }
#pragma unroll
  for (int n = 0; n < 16; ++n) {
    float bias = b2b[n * 16 + l16];
#pragma unroll
    for (int r = 0; r < 8; ++r) {
      float v = fmaxf(acc3[n][r] + bias, 0.0f);
      out[((size_t)b * N_ + (p0 + w * 16 + r + 8 * half)) * C_ + n * 16 + l16] = v;
    }
  }
}

// ---------------------------------------------------------------------------
// Kernels 5a/5b: deterministic two-stage mean over N (no atomics -> graph
// replays are bit-stable). 256 partial blocks to use HBM bandwidth.
// ---------------------------------------------------------------------------
__global__ __launch_bounds__(256)
void gpart_kernel(const float* __restrict__ F, float* __restrict__ gp) {
  const int b = blockIdx.x / NCHUNK, ch = blockIdx.x % NCHUNK, c = threadIdx.x;
  const int rows = N_ / NCHUNK;   // 512
  float s = 0.0f;
  const float* base = F + ((size_t)b * N_ + (size_t)ch * rows) * C_ + c;
  for (int n = 0; n < rows; ++n) s += base[(size_t)n * C_];
  gp[(b * NCHUNK + ch) * C_ + c] = s;
}

__global__ __launch_bounds__(256)
void gfinal_kernel(const float* __restrict__ gp, float* __restrict__ g) {
  const int b = blockIdx.x, c = threadIdx.x;
  float s = 0.0f;
  for (int ch = 0; ch < NCHUNK; ++ch) s += gp[(b * NCHUNK + ch) * C_ + c];
  g[b * C_ + c] = s * (1.0f / (float)N_);
}

// ---------------------------------------------------------------------------
extern "C" void kernel_launch(void* const* d_in, const int* in_sizes, int n_in,
                              void* d_out, int out_size, void* d_ws, size_t ws_size,
                              hipStream_t stream) {
  (void)in_sizes; (void)n_in; (void)out_size; (void)ws_size;
  const float* pts = (const float*)d_in[0];
  const float* W1a = (const float*)d_in[1];
  const float* b1a = (const float*)d_in[2];
  const float* W1b = (const float*)d_in[3];
  const float* b1b = (const float*)d_in[4];
  const float* W2a = (const float*)d_in[5];
  const float* b2a = (const float*)d_in[6];
  const float* W2b = (const float*)d_in[7];
  const float* b2b = (const float*)d_in[8];

  // workspace layout (bytes): ~1.5 MB total
  char* ws = (char*)d_ws;
  int*    idx     = (int*)(ws + 0);                        // 8*512*4      = 16384
  float*  sxyz    = (float*)(ws + 16384);                  // 8*3*512*4    = 49152
  __bf16* anch_cm = (__bf16*)(ws + 65536);                 // 8*128*512*2  = 1048576
  __bf16* w2a_cm  = (__bf16*)(ws + 65536 + 1048576);       // 256*128*2    = 65536
  __bf16* w2b_cm  = (__bf16*)(ws + 65536 + 1048576 + 65536);         // 256*256*2   = 131072
  float*  gpart   = (float*)(ws + 65536 + 1048576 + 65536 + 131072); // 8*32*256*4  = 262144

  float* F = (float*)d_out;                        // [8,16384,256]
  float* g = F + (size_t)B_ * N_ * C_;             // [8,256]

  fps_kernel<<<B_, 1024, 0, stream>>>(pts, idx);
  prep_weights<<<(C_ * C_) / 256, 256, 0, stream>>>(W2a, W2b, w2a_cm, w2b_cm);
  anchors_kernel<<<B_ * M_, H_, 0, stream>>>(pts, idx, W1a, b1a, W1b, b1b, anch_cm, sxyz);
  fused_kernel<<<B_ * (N_ / PTILE), 256, 0, stream>>>(pts, anch_cm, sxyz,
                                                      w2a_cm, b2a, w2b_cm, b2b, F);
  gpart_kernel<<<B_ * NCHUNK, 256, 0, stream>>>(F, gpart);
  gfinal_kernel<<<B_, 256, 0, stream>>>(gpart, g);
}
